// MPMoE_33689723469988
// MI455X (gfx1250) — matrix operand order repaired and checked by reading, not compiled, not run on hardware
//
#include <hip/hip_runtime.h>

#define B_SZ 32768
#define D_SZ 512
#define E_SZ 64
#define K_TOP 4
#define H_SZ 5376          // (16/2 - 1) * 768
#define NOISE_EPS 0.01f
#define LOSS_COEF 0.01f

#define WAVES_PER_BLOCK 4
#define ROWS_PER_WAVE   16
#define MAIN_BLOCKS (B_SZ / (WAVES_PER_BLOCK * ROWS_PER_WAVE))   // 512

typedef __attribute__((ext_vector_type(16))) _Float16 v16h;
typedef __attribute__((ext_vector_type(8)))  float    v8f;
typedef __attribute__((ext_vector_type(4)))  float    v4f;   // native vector for NT ld/st

// ---------------------------------------------------------------------------
// Kernel 1: pack w_gate / w_noise (fp32, DxE row-major) into f16 WMMA
// B-fragment layout. For v_wmma_f32_16x16x32_f16, the 32x16 B tile is held as
// v16h per lane: lane group g = lane>>4 covers K = (h>>3)*16 + g*8 + (h&7) (h = half index
// 0..15), column N = t*16 + (lane&15).  Packed element index:
//   ((kb*4 + t)*32 + lane)*16 + h
// so the main kernel loads one contiguous 32-byte v16h per lane per tile.
// Also zeroes the 128-float stats accumulator (importance[64] ++ load[64]).
// ---------------------------------------------------------------------------
__global__ void pack_weights(const float* __restrict__ wg,
                             const float* __restrict__ wn,
                             _Float16* __restrict__ pg,
                             _Float16* __restrict__ pn,
                             float* __restrict__ stats) {
    int idx = blockIdx.x * blockDim.x + threadIdx.x;     // 0 .. 65535
    if (blockIdx.x == 0 && threadIdx.x < 128) stats[threadIdx.x] = 0.f;
    int which = idx >> 15;                               // 0 = gate, 1 = noise
    int i  = idx & 32767;
    int h  = i & 15;
    int l  = (i >> 4) & 31;
    int t  = (i >> 9) & 3;
    int kb = i >> 11;
    int k  = kb * 32 + (h >> 3) * 16 + (l >> 4) * 8 + (h & 7);
    int n  = t * 16 + (l & 15);
    const float* src = which ? wn : wg;
    _Float16*    dst = which ? pn : pg;
    dst[i] = (_Float16)src[(size_t)k * E_SZ + n];
}

// ---------------------------------------------------------------------------
// Kernel 2: fused gating GEMMs (WMMA f16->f32) + softplus + noisy top-5 +
// softmax(top-4) + importance/load accumulation + sparse dispatch einsum.
// One wave per 16-row tile; 4 waves / block; 512 blocks.
// ---------------------------------------------------------------------------
__global__ void __launch_bounds__(WAVES_PER_BLOCK * 32)
moe_main(const float* __restrict__ x,
         const float* __restrict__ noise,
         const float* __restrict__ ctx,
         const _Float16* __restrict__ pg,
         const _Float16* __restrict__ pn,
         float* __restrict__ stats,
         float* __restrict__ out) {
    __shared__ float s_clean[WAVES_PER_BLOCK][ROWS_PER_WAVE][E_SZ];
    __shared__ float s_std  [WAVES_PER_BLOCK][ROWS_PER_WAVE][E_SZ];
    __shared__ float s_imp[E_SZ];
    __shared__ float s_load[E_SZ];

    const int tid  = threadIdx.x;
    const int wv   = tid >> 5;
    const int lane = tid & 31;
    if (tid < E_SZ) { s_imp[tid] = 0.f; s_load[tid] = 0.f; }
    __syncthreads();

    const int row0 = (blockIdx.x * WAVES_PER_BLOCK + wv) * ROWS_PER_WAVE;

    // ---------------- Phase 1: WMMA gating GEMMs over D=512 ----------------
    const v8f zacc = {0.f, 0.f, 0.f, 0.f, 0.f, 0.f, 0.f, 0.f};
    v8f accg[4], accn[4];
#pragma unroll
    for (int t = 0; t < 4; ++t) { accg[t] = zacc; accn[t] = zacc; }

    // 16-bit A (16x32) layout: lane m=lane&15 is row M=m; lane group 0 holds
    // K pairs 0..7 then 16..23 in halves 0..7 / 8..15; group 1 holds 8..15 /
    // 24..31. koff = (lane>>4)*8 selects the group's K base.
    const int m    = lane & 15;
    const int koff = (lane >> 4) * 8;
    const float* xrow = x + (size_t)(row0 + m) * D_SZ;

    for (int kb = 0; kb < D_SZ / 32; ++kb) {
        const float* xr = xrow + kb * 32;
        v4f f0 = __builtin_nontemporal_load((const v4f*)(xr + koff));
        v4f f1 = __builtin_nontemporal_load((const v4f*)(xr + koff + 4));
        v4f f2 = __builtin_nontemporal_load((const v4f*)(xr + koff + 16));
        v4f f3 = __builtin_nontemporal_load((const v4f*)(xr + koff + 20));
        v16h a;
        a[0]  = (_Float16)f0.x; a[1]  = (_Float16)f0.y;
        a[2]  = (_Float16)f0.z; a[3]  = (_Float16)f0.w;
        a[4]  = (_Float16)f1.x; a[5]  = (_Float16)f1.y;
        a[6]  = (_Float16)f1.z; a[7]  = (_Float16)f1.w;
        a[8]  = (_Float16)f2.x; a[9]  = (_Float16)f2.y;
        a[10] = (_Float16)f2.z; a[11] = (_Float16)f2.w;
        a[12] = (_Float16)f3.x; a[13] = (_Float16)f3.y;
        a[14] = (_Float16)f3.z; a[15] = (_Float16)f3.w;
#pragma unroll
        for (int t = 0; t < 4; ++t) {
            const size_t boff = ((size_t)(kb * 4 + t) * 32 + lane) * 16;
            v16h bg = *(const v16h*)(pg + boff);
            accg[t] = __builtin_amdgcn_wmma_f32_16x16x32_f16(
                false, a, false, bg, (short)0, accg[t], false, false);
            v16h bn = *(const v16h*)(pn + boff);
            accn[t] = __builtin_amdgcn_wmma_f32_16x16x32_f16(
                false, a, false, bn, (short)0, accn[t], false, false);
        }
    }

    // Spill C tiles to wave-private LDS: VGPR r -> M = r (lanes 0-15) or 8+r
    // (lanes 16-31), N = t*16 + (lane&15). Noise path gets softplus + eps.
    const int ml = (lane < 16) ? 0 : 8;
    const int nl = lane & 15;
#pragma unroll
    for (int t = 0; t < 4; ++t) {
#pragma unroll
        for (int r = 0; r < 8; ++r) {
            float cg = accg[t][r];
            float cn = accn[t][r];
            float sp = (cn > 20.f) ? cn : log1pf(__expf(cn));
            s_clean[wv][ml + r][t * 16 + nl] = cg;
            s_std  [wv][ml + r][t * 16 + nl] = sp + NOISE_EPS;
        }
    }
    // Wave-private LDS region: same-wave store->load ordering is handled by
    // compiler-inserted s_wait_dscnt; no cross-wave barrier required here.

    // ---------------- Phase 2: per-row top-k, gates, stats, dispatch -------
    float loadacc0 = 0.f, loadacc1 = 0.f;           // e = lane, lane+32
    for (int r = 0; r < ROWS_PER_WAVE; ++r) {
        const int row = row0 + r;
        float c0  = s_clean[wv][r][lane];
        float c1  = s_clean[wv][r][lane + 32];
        float sd0 = s_std[wv][r][lane];
        float sd1 = s_std[wv][r][lane + 32];
        float nz0 = __builtin_nontemporal_load(noise + (size_t)row * E_SZ + lane);
        float nz1 = __builtin_nontemporal_load(noise + (size_t)row * E_SZ + lane + 32);
        float ny0 = c0 + nz0 * sd0;
        float ny1 = c1 + nz1 * sd1;

        // top-5 by wave butterfly argmax, ties broken toward lower index
        bool rem0 = false, rem1 = false;
        float tv[5]; int ti[5];
#pragma unroll
        for (int j = 0; j < 5; ++j) {
            float v  = rem0 ? -3.0e38f : ny0;
            int   id = lane;
            float v1 = rem1 ? -3.0e38f : ny1;
            if (v1 > v) { v = v1; id = lane + 32; }
#pragma unroll
            for (int off = 16; off > 0; off >>= 1) {
                float ov = __shfl_xor(v, off, 32);
                int   oi = __shfl_xor(id, off, 32);
                if (ov > v || (ov == v && oi < id)) { v = ov; id = oi; }
            }
            tv[j] = v; ti[j] = id;
            if (id == lane)           rem0 = true;
            else if (id == lane + 32) rem1 = true;
        }

        // softmax over the top-4 noisy logits
        float e0 = __expf(tv[0] - tv[0]);
        float e1 = __expf(tv[1] - tv[0]);
        float e2 = __expf(tv[2] - tv[0]);
        float e3 = __expf(tv[3] - tv[0]);
        float inv = 1.f / (e0 + e1 + e2 + e3);
        float g0 = e0 * inv, g1 = e1 * inv, g2 = e2 * inv, g3 = e3 * inv;

        if (lane == 0) {
            atomicAdd(&s_imp[ti[0]], g0);
            atomicAdd(&s_imp[ti[1]], g1);
            atomicAdd(&s_imp[ti[2]], g2);
            atomicAdd(&s_imp[ti[3]], g3);
        }

        // load: Phi((clean - thr) / std), thr depends on membership in top-4
        const float thrIn = tv[4], thrOut = tv[3];
        {
            float th0 = (ny0 > thrIn) ? thrIn : thrOut;
            loadacc0 += 0.5f * (1.f + erff((c0 - th0) / sd0 * 0.70710678f));
            float th1 = (ny1 > thrIn) ? thrIn : thrOut;
            loadacc1 += 0.5f * (1.f + erff((c1 - th1) / sd1 * 0.70710678f));
        }

        // dispatch: out[row,:] = sum_k g_k * ctx_c[idx_k,:]  (ctx_c L2-hot,
        // out streamed with non-temporal 128-bit stores)
        const v4f* r0p = (const v4f*)(ctx + (size_t)ti[0] * H_SZ);
        const v4f* r1p = (const v4f*)(ctx + (size_t)ti[1] * H_SZ);
        const v4f* r2p = (const v4f*)(ctx + (size_t)ti[2] * H_SZ);
        const v4f* r3p = (const v4f*)(ctx + (size_t)ti[3] * H_SZ);
        v4f* op = (v4f*)(out + (size_t)row * H_SZ);
        for (int it = lane; it < H_SZ / 4; it += 32) {
            v4f a0 = r0p[it], a1 = r1p[it], a2 = r2p[it], a3 = r3p[it];
            v4f o = g0 * a0 + g1 * a1 + g2 * a2 + g3 * a3;
            __builtin_nontemporal_store(o, op + it);
        }
    }
    atomicAdd(&s_load[lane],      loadacc0);
    atomicAdd(&s_load[lane + 32], loadacc1);
    __syncthreads();
    if (tid < E_SZ) {
        atomicAdd(&stats[tid],         s_imp[tid]);
        atomicAdd(&stats[E_SZ + tid],  s_load[tid]);
    }
}

// ---------------------------------------------------------------------------
// Kernel 3: aux_loss = (cv^2(importance) + cv^2(load)) * LOSS_COEF
// ---------------------------------------------------------------------------
__global__ void finalize_aux(const float* __restrict__ stats,
                             float* __restrict__ auxp) {
    if (threadIdx.x == 0 && blockIdx.x == 0) {
        float si = 0.f, si2 = 0.f, sl = 0.f, sl2 = 0.f;
        for (int i = 0; i < E_SZ; ++i) {
            float a = stats[i];        si += a; si2 += a * a;
            float b = stats[E_SZ + i]; sl += b; sl2 += b * b;
        }
        float mi = si / (float)E_SZ;
        float vi = si2 / (float)E_SZ - mi * mi;
        float ml = sl / (float)E_SZ;
        float vl = sl2 / (float)E_SZ - ml * ml;
        float cvi = vi / (mi * mi + 1e-10f);
        float cvl = vl / (ml * ml + 1e-10f);
        auxp[0] = (cvi + cvl) * LOSS_COEF;
    }
}

// ---------------------------------------------------------------------------
extern "C" void kernel_launch(void* const* d_in, const int* in_sizes, int n_in,
                              void* d_out, int out_size, void* d_ws, size_t ws_size,
                              hipStream_t stream) {
    (void)in_sizes; (void)n_in; (void)out_size; (void)ws_size;
    const float* x       = (const float*)d_in[0];
    const float* w_gate  = (const float*)d_in[1];
    const float* w_noise = (const float*)d_in[2];
    const float* noise   = (const float*)d_in[3];
    const float* ctx     = (const float*)d_in[4];
    float* out = (float*)d_out;

    char* ws = (char*)d_ws;
    _Float16* pg    = (_Float16*)(ws + 0);            // 64 KB packed w_gate
    _Float16* pn    = (_Float16*)(ws + 65536);        // 64 KB packed w_noise
    float*    stats = (float*)(ws + 131072);          // importance[64] ++ load[64]

    pack_weights<<<256, 256, 0, stream>>>(w_gate, w_noise, pg, pn, stats);
    moe_main<<<MAIN_BLOCKS, WAVES_PER_BLOCK * 32, 0, stream>>>(
        x, noise, ctx, pg, pn, stats, out);
    finalize_aux<<<1, 32, 0, stream>>>(stats, out + (size_t)B_SZ * H_SZ);
}
